// MLATorch_25795573579937
// MI455X (gfx1250) — matrix operand order repaired and checked by reading, not compiled, not run on hardware
//
#include <hip/hip_runtime.h>
#include <hip/hip_bf16.h>

typedef __attribute__((ext_vector_type(16))) _Float16 v16h;
typedef __attribute__((ext_vector_type(8)))  _Float16 v8h;
typedef __attribute__((ext_vector_type(8)))  float    v8f;
typedef unsigned int v4u __attribute__((ext_vector_type(4)));
typedef int          v8i __attribute__((ext_vector_type(8)));
typedef int          v4i __attribute__((ext_vector_type(4)));

#ifdef __has_builtin
#if __has_builtin(__builtin_amdgcn_tensor_load_to_lds)
#define HAVE_TDM 1
#endif
#endif

namespace {
constexpr int BATCH = 8, SEQ = 32, TOK = BATCH * SEQ;      // 256 new tokens
constexpr int DIM = 2048, KV_RANK = 512, Q_RANK = 768;
constexpr int NH = 16, D_NR = 128, D_R = 64, D_V = 128;
constexpr int QK = D_NR + D_R;           // 192
constexpr int CF = KV_RANK + D_R;        // 576 (latent + rope concat)
constexpr int MAX_T = 4096, START = MAX_T - SEQ;           // 4064
constexpr float RMS_EPS = 1.1920929e-07f;
constexpr float SCALE = 0.07216878364870322f;              // 1/sqrt(192)

// ---- workspace layout (float offsets) ----
constexpr size_t WS_QDOWN = 0;                               // 256*768
constexpr size_t WS_QNORM = WS_QDOWN + (size_t)TOK * Q_RANK;
constexpr size_t WS_Q     = WS_QNORM + (size_t)TOK * Q_RANK; // 256*3072
constexpr size_t WS_KV    = WS_Q     + (size_t)TOK * NH * QK; // 256*576
constexpr size_t WS_QFULL = WS_KV    + (size_t)TOK * CF;      // 256*16*576
constexpr size_t WS_O     = WS_QFULL + (size_t)TOK * NH * CF; // 256*16*512
constexpr size_t WS_OHEAD = WS_O     + (size_t)TOK * NH * KV_RANK; // 256*2048
constexpr size_t WS_END_F32 = WS_OHEAD + (size_t)TOK * NH * D_V;
constexpr size_t WS_S16_BYTE = WS_END_F32 * sizeof(float);   // 4096*4096 f16
} // namespace

// ---------------------------------------------------------------------------
// Raw (f32) fragment loads + conversion. contiguous k order used by this file (not the ISA table order), wave32:
// A (16-bit 16x32): lane<16 -> halves K0..15; lane>=16 -> K16..31
// B (32x16): lane = N col; halves = K (hi*16+e)
// ---------------------------------------------------------------------------
__device__ __forceinline__ void ldA_raw(const float* p, int k0, int hi, float4* q) {
  q[0] = *(const float4*)(p + k0 + hi * 16);
  q[1] = *(const float4*)(p + k0 + hi * 16 + 4);
  q[2] = *(const float4*)(p + k0 + hi * 16 + 8);
  q[3] = *(const float4*)(p + k0 + hi * 16 + 12);
}

__device__ __forceinline__ void ldB_raw(const float* p, int k0, int hi, float4* q) {
  q[0] = *(const float4*)(p + k0 + hi * 16);
  q[1] = *(const float4*)(p + k0 + hi * 16 + 4);
  q[2] = *(const float4*)(p + k0 + hi * 16 + 8);
  q[3] = *(const float4*)(p + k0 + hi * 16 + 12);
}

__device__ __forceinline__ v16h cvt16(const float4* q) {
  v16h r;
#pragma unroll
  for (int i = 0; i < 4; ++i) {
    r[4 * i + 0] = (_Float16)q[i].x;
    r[4 * i + 1] = (_Float16)q[i].y;
    r[4 * i + 2] = (_Float16)q[i].z;
    r[4 * i + 3] = (_Float16)q[i].w;
  }
  return r;
}

__device__ __forceinline__ v16h packB_f16(const _Float16* p, int k0, int hi) {
  const v8h lo = *(const v8h*)(p + k0 + hi * 16);
  const v8h hh = *(const v8h*)(p + k0 + hi * 16 + 8);
  v16h r;
#pragma unroll
  for (int e = 0; e < 8; ++e) { r[e] = lo[e]; r[e + 8] = hh[e]; }
  return r;
}

__device__ __forceinline__ v16h joinA_f16(const v8h lo, const v8h hh) {
  v16h r;
#pragma unroll
  for (int e = 0; e < 8; ++e) { r[e] = lo[e]; r[e + 8] = hh[e]; }
  return r;
}

#define WMMA(a, b, c) \
  __builtin_amdgcn_wmma_f32_16x16x32_f16(false, (a), false, (b), (short)0, (c), false, false)

// ---------------------------------------------------------------------------
// One wave computes a 16x64 tile, software-pipelined double buffer:
// per iteration = 2 k-steps of 32, 8 wmma. K must be a multiple of 64.
// ---------------------------------------------------------------------------
template <typename RA, typename RB, typename ST>
__device__ __forceinline__ void gemm16x64(int K, RA rowA, RB rowB, ST st) {
  const int lane = threadIdx.x & 31;
  const int hi = lane >> 4, li = lane & 15;
  const float* pa = rowA(li);
  const float* pb[4];
#pragma unroll
  for (int nf = 0; nf < 4; ++nf) pb[nf] = rowB(nf * 16 + li);
  v8f acc[4] = {};
  float4 ra0[4], ra1[4], rb0[4][4], rb1[4][4];
  ldA_raw(pa, 0, hi, ra0);
#pragma unroll
  for (int nf = 0; nf < 4; ++nf) ldB_raw(pb[nf], 0, hi, rb0[nf]);
  for (int k0 = 0; k0 < K; k0 += 64) {
    // prefetch second half-step
    ldA_raw(pa, k0 + 32, hi, ra1);
#pragma unroll
    for (int nf = 0; nf < 4; ++nf) ldB_raw(pb[nf], k0 + 32, hi, rb1[nf]);
    // compute first half-step
    {
      const v16h a = cvt16(ra0);
#pragma unroll
      for (int nf = 0; nf < 4; ++nf) acc[nf] = WMMA(a, cvt16(rb0[nf]), acc[nf]);
    }
    // prefetch next first half-step
    if (k0 + 64 < K) {
      ldA_raw(pa, k0 + 64, hi, ra0);
#pragma unroll
      for (int nf = 0; nf < 4; ++nf) ldB_raw(pb[nf], k0 + 64, hi, rb0[nf]);
    }
    // compute second half-step
    {
      const v16h a = cvt16(ra1);
#pragma unroll
      for (int nf = 0; nf < 4; ++nf) acc[nf] = WMMA(a, cvt16(rb1[nf]), acc[nf]);
    }
  }
#pragma unroll
  for (int nf = 0; nf < 4; ++nf)
#pragma unroll
    for (int r = 0; r < 8; ++r) st(hi * 8 + r, nf * 16 + li, acc[nf][r]);
}

__device__ __forceinline__ int global_wave() {
  return (int)((blockIdx.x * blockDim.x + threadIdx.x) >> 5);
}

// ---------------------------------------------------------------------------
// K1: fused down projection  x(256x2048) x [w_q_down;w_kv_down]^T -> qdown|kv
// ---------------------------------------------------------------------------
__global__ void __launch_bounds__(256)
k_down_proj(const float* __restrict__ x, const float* __restrict__ wq,
            const float* __restrict__ wkv, float* __restrict__ qdown,
            float* __restrict__ kv) {
  const int gw = global_wave();            // 336 waves: 16 m x 21 ngroups
  const int ng = gw % 21, mtile = gw / 21;
  const int m0 = mtile * 16, n0 = ng * 64;
  gemm16x64(DIM,
    [&](int m) { return x + (size_t)(m0 + m) * DIM; },
    [&](int n) {
      const int nn = n0 + n;
      return nn < Q_RANK ? wq + (size_t)nn * DIM : wkv + (size_t)(nn - Q_RANK) * DIM;
    },
    [&](int m, int n, float v) {
      const int nn = n0 + n, mm = m0 + m;
      if (nn < Q_RANK) qdown[mm * Q_RANK + nn] = v;
      else             kv[mm * CF + (nn - Q_RANK)] = v;
    });
}

// ---------------------------------------------------------------------------
// K2: RMSNorm(q_down) + RoPE on k_rope (in place in ws kv)
// ---------------------------------------------------------------------------
__global__ void __launch_bounds__(256)
k_rms_krope(const float* __restrict__ qdown, const float* __restrict__ rmsw,
            const float* __restrict__ fc, const float* __restrict__ fs,
            float* __restrict__ qnorm, float* __restrict__ kv) {
  const int m = blockIdx.x;
  const int l = m & (SEQ - 1);
  __shared__ float red[256];
  float s = 0.f;
  for (int j = threadIdx.x; j < Q_RANK; j += 256) {
    const float v = qdown[m * Q_RANK + j];
    s += v * v;
  }
  red[threadIdx.x] = s;
  __syncthreads();
  for (int st = 128; st > 0; st >>= 1) {
    if ((int)threadIdx.x < st) red[threadIdx.x] += red[threadIdx.x + st];
    __syncthreads();
  }
  const float inv = rsqrtf(red[0] / (float)Q_RANK + RMS_EPS);
  for (int j = threadIdx.x; j < Q_RANK; j += 256)
    qnorm[m * Q_RANK + j] = qdown[m * Q_RANK + j] * inv * rmsw[j];
  if (threadIdx.x < D_R / 2) {
    const int i = threadIdx.x;
    const float c = fc[l * (D_R / 2) + i], sn = fs[l * (D_R / 2) + i];
    float* kr = kv + m * CF + KV_RANK;
    const float xr = kr[2 * i], xi = kr[2 * i + 1];
    kr[2 * i]     = xr * c - xi * sn;
    kr[2 * i + 1] = xr * sn + xi * c;
  }
}

// ---------------------------------------------------------------------------
// K3: q up-projection  qnorm(256x768) x w_q_up^T -> q(256x3072)
// ---------------------------------------------------------------------------
__global__ void __launch_bounds__(256)
k_q_up(const float* __restrict__ qnorm, const float* __restrict__ wqu,
       float* __restrict__ q) {
  const int gw = global_wave();            // 768 waves: 16 m x 48 ngroups
  const int ng = gw % 48, mtile = gw / 48;
  const int m0 = mtile * 16, n0 = ng * 64;
  gemm16x64(Q_RANK,
    [&](int m) { return qnorm + (size_t)(m0 + m) * Q_RANK; },
    [&](int n) { return wqu + (size_t)(n0 + n) * Q_RANK; },
    [&](int m, int n, float v) { q[(m0 + m) * (NH * QK) + n0 + n] = v; });
}

// ---------------------------------------------------------------------------
// K4: RoPE on q_rope, written into q_full[..., 512:576]
// ---------------------------------------------------------------------------
__global__ void __launch_bounds__(256)
k_qrope(const float* __restrict__ q, const float* __restrict__ fc,
        const float* __restrict__ fs, float* __restrict__ qfull) {
  const int idx = blockIdx.x * blockDim.x + threadIdx.x; // TOK*NH*32
  const int i = idx & 31;
  const int h = (idx >> 5) & (NH - 1);
  const int m = idx >> 9;
  const int l = m & (SEQ - 1);
  const float* qr = q + m * (NH * QK) + h * QK + D_NR;
  const float c = fc[l * (D_R / 2) + i], sn = fs[l * (D_R / 2) + i];
  const float xr = qr[2 * i], xi = qr[2 * i + 1];
  float* dst = qfull + ((size_t)(m * NH + h)) * CF + KV_RANK;
  dst[2 * i]     = xr * c - xi * sn;
  dst[2 * i + 1] = xr * sn + xi * c;
}

// ---------------------------------------------------------------------------
// K5: q_abs per head: q_nrope(256x128) x w_up_nr[h](128x512) -> qfull[...,0:512]
// W columns have stride 512 (K slow axis); K=128 only -> gather B (cold kernel).
// ---------------------------------------------------------------------------
__global__ void __launch_bounds__(256)
k_q_abs(const float* __restrict__ q, const float* __restrict__ wkvu,
        float* __restrict__ qfull) {
  const int gw = global_wave();            // 2048 waves: 16 h x 16 m x 8 ng
  const int h = gw / 128;
  const int rem = gw % 128;
  const int mtile = rem / 8, ng = rem % 8;
  const int m0 = mtile * 16, n0 = ng * 64;
  const int lane = threadIdx.x & 31;
  const int hi = lane >> 4, li = lane & 15;
  const float* pa = q + (size_t)(m0 + li) * (NH * QK) + h * QK;
  v8f acc[4] = {};
  for (int k0 = 0; k0 < D_NR; k0 += 32) {
    float4 ra[4];
    ldA_raw(pa, k0, hi, ra);
    const v16h a = cvt16(ra);
#pragma unroll
    for (int nf = 0; nf < 4; ++nf) {
      const int col = n0 + nf * 16 + li;
      const float* pb = wkvu + (size_t)(h * (D_NR + D_V)) * KV_RANK + col;
      v16h b;
#pragma unroll
      for (int e = 0; e < 16; ++e)
        b[e] = (_Float16)pb[(size_t)(k0 + hi * 16 + e) * KV_RANK];
      acc[nf] = WMMA(a, b, acc[nf]);
    }
  }
#pragma unroll
  for (int nf = 0; nf < 4; ++nf)
#pragma unroll
    for (int r = 0; r < 8; ++r)
      qfull[((size_t)((m0 + hi * 8 + r) * NH + h)) * CF + n0 + nf * 16 + li] = acc[nf][r];
}

// ---------------------------------------------------------------------------
// K6: scores  S[bh](32x4096) = q_full(32x576) x C(4096x576)^T  (f16 out)
// Q tile staged to LDS via Tensor Data Mover; B double-buffered from caches.
// ---------------------------------------------------------------------------
__global__ void __launch_bounds__(128)
k_scores(const float* __restrict__ qfull, const float* __restrict__ kvcache,
         const float* __restrict__ krcache, const float* __restrict__ kvnew,
         _Float16* __restrict__ S) {
  const int blk = blockIdx.x;              // 4096 = 128 bh x 2 m x 16 chunk
  const int chunk = blk & 15;
  const int mtile = (blk >> 4) & 1;
  const int bh = blk >> 5;
  const int b = bh >> 4, h = bh & 15;
  const int l0 = mtile * 16;
  __shared__ float qtile[16 * CF];         // 36 KB

  const float* gsrc = qfull + ((size_t)((b * SEQ + l0) * NH + h)) * CF;
#ifdef HAVE_TDM
  if (threadIdx.x < 32) {
    const unsigned long long ga = (unsigned long long)gsrc;
    const unsigned lds = (unsigned)(unsigned long long)(&qtile[0]);
    v4u g0;
    g0[0] = 1u;                                   // count=1, user descriptor
    g0[1] = lds;                                  // lds_addr
    g0[2] = (unsigned)ga;                         // global_addr[31:0]
    g0[3] = (unsigned)(ga >> 32) | (2u << 30);    // global_addr[56:32] | type=2
    v8i g1;
    g1[0] = (int)(2u << 16);                      // data_size = 4B
    g1[1] = (int)((unsigned)CF << 16);            // tensor_dim0[15:0] @ [31:16]
    g1[2] = (int)(16u << 16);                     // tensor_dim1[15:0] = 16 rows
    g1[3] = (int)((unsigned)CF << 16);            // tile_dim0 = 576 @ [127:112]
    g1[4] = 16;                                   // tile_dim1 = 16
    g1[5] = NH * CF;                              // tensor_dim0_stride (elems)
    g1[6] = 0;
    g1[7] = 0;
    const v4i gz4 = {0, 0, 0, 0};
    const v8i gz8 = {0, 0, 0, 0, 0, 0, 0, 0};
    __builtin_amdgcn_tensor_load_to_lds(g0, g1, gz4, gz4, gz8, 0);
    __builtin_amdgcn_s_wait_tensorcnt(0);
  }
#else
  for (int i = threadIdx.x; i < 16 * CF / 4; i += 128) {
    const int r = i / (CF / 4), c = i % (CF / 4);
    *(float4*)&qtile[r * CF + c * 4] =
        *(const float4*)&gsrc[(size_t)r * NH * CF + c * 4];
  }
#endif
  __syncthreads();

  const int lane = threadIdx.x & 31;
  const int hi = lane >> 4, li = lane & 15;
  const int ng = threadIdx.x >> 5;         // 4 waves: 64 cols each
  const int t0 = chunk * 256 + ng * 64;

  const float* pl[4];
  const float* pr[4];
#pragma unroll
  for (int nf = 0; nf < 4; ++nf) {
    const int t = t0 + nf * 16 + li;
    if (t < START) {
      pl[nf] = kvcache + ((size_t)b * MAX_T + t) * KV_RANK;
      pr[nf] = krcache + ((size_t)b * MAX_T + t) * D_R - KV_RANK;
    } else {
      const float* pn = kvnew + (size_t)(b * SEQ + (t - START)) * CF;
      pl[nf] = pn;
      pr[nf] = pn;
    }
  }
  const float* pa = qtile + li * CF;
  auto bp = [&](int nf, int k) { return k < KV_RANK ? pl[nf] : pr[nf]; };

  v8f acc[4] = {};
  float4 ra0[4], ra1[4], rb0[4][4], rb1[4][4];
  ldA_raw(pa, 0, hi, ra0);
#pragma unroll
  for (int nf = 0; nf < 4; ++nf) ldB_raw(bp(nf, 0), 0, hi, rb0[nf]);
  for (int k0 = 0; k0 < CF; k0 += 64) {        // CF = 9 * 64
    ldA_raw(pa, k0 + 32, hi, ra1);
#pragma unroll
    for (int nf = 0; nf < 4; ++nf) ldB_raw(bp(nf, k0 + 32), k0 + 32, hi, rb1[nf]);
    {
      const v16h a = cvt16(ra0);
#pragma unroll
      for (int nf = 0; nf < 4; ++nf) acc[nf] = WMMA(a, cvt16(rb0[nf]), acc[nf]);
    }
    if (k0 + 64 < CF) {
      ldA_raw(pa, k0 + 64, hi, ra0);
#pragma unroll
      for (int nf = 0; nf < 4; ++nf) ldB_raw(bp(nf, k0 + 64), k0 + 64, hi, rb0[nf]);
    }
    {
      const v16h a = cvt16(ra1);
#pragma unroll
      for (int nf = 0; nf < 4; ++nf) acc[nf] = WMMA(a, cvt16(rb1[nf]), acc[nf]);
    }
  }
#pragma unroll
  for (int nf = 0; nf < 4; ++nf)
#pragma unroll
    for (int r = 0; r < 8; ++r)
      S[((size_t)(bh * SEQ + l0 + hi * 8 + r)) * MAX_T + t0 + nf * 16 + li] =
          (_Float16)acc[nf][r];
}

// ---------------------------------------------------------------------------
// K7: masked softmax, in place on the f16 score buffer (one block per row)
// ---------------------------------------------------------------------------
__global__ void __launch_bounds__(256)
k_softmax(_Float16* __restrict__ S) {
  const int row = blockIdx.x;              // B*H*SEQ = 4096 rows
  const int l = row & (SEQ - 1);
  const int limit = START + l;
  __shared__ float red[256];
  float vals[MAX_T / 256];
  float mx = -1e30f;
#pragma unroll
  for (int j = 0; j < MAX_T / 256; ++j) {
    const int t = j * 256 + threadIdx.x;
    const float v = (float)S[(size_t)row * MAX_T + t] * SCALE;
    vals[j] = v;
    if (t <= limit && v > mx) mx = v;
  }
  red[threadIdx.x] = mx;
  __syncthreads();
  for (int st = 128; st > 0; st >>= 1) {
    if ((int)threadIdx.x < st)
      red[threadIdx.x] = fmaxf(red[threadIdx.x], red[threadIdx.x + st]);
    __syncthreads();
  }
  mx = red[0];
  __syncthreads();
  float sum = 0.f;
#pragma unroll
  for (int j = 0; j < MAX_T / 256; ++j) {
    const int t = j * 256 + threadIdx.x;
    const float e = (t <= limit) ? __expf(vals[j] - mx) : 0.f;
    vals[j] = e;
    sum += e;
  }
  red[threadIdx.x] = sum;
  __syncthreads();
  for (int st = 128; st > 0; st >>= 1) {
    if ((int)threadIdx.x < st) red[threadIdx.x] += red[threadIdx.x + st];
    __syncthreads();
  }
  const float inv = 1.f / red[0];
#pragma unroll
  for (int j = 0; j < MAX_T / 256; ++j) {
    const int t = j * 256 + threadIdx.x;
    S[(size_t)row * MAX_T + t] = (_Float16)(vals[j] * inv);
  }
}

// ---------------------------------------------------------------------------
// K8: O[bh](32x512) = P(32x4096 f16) x Ckv(4096x512)
// Ckv staged transposed in LDS ([feature][t], pad 8) per 64-t chunk.
// ---------------------------------------------------------------------------
__global__ void __launch_bounds__(256)
k_pv(const _Float16* __restrict__ P, const float* __restrict__ kvcache,
     const float* __restrict__ kvnew, float* __restrict__ O) {
  constexpr int KT = 64, CTP = KT + 8;     // padded LDS stride (halves)
  const int blk = blockIdx.x;              // 256 = 128 bh x 2 nchunk
  const int nc = blk & 1;
  const int bh = blk >> 1;
  const int b = bh >> 4, h = bh & 15;
  const int n0b = nc * 256;
  __shared__ _Float16 ct[256 * CTP];       // 36 KB

  const int lane = threadIdx.x & 31;
  const int hi = lane >> 4, li = lane & 15;
  const int w = threadIdx.x >> 5;          // 8 waves
  const int mtile = w & 1, ng = w >> 1;    // 2 m-tiles x 4 n-groups

  const _Float16* pa =
      P + ((size_t)(bh * SEQ + mtile * 16 + li)) * MAX_T;
  v8f acc[4] = {};

  for (int t0 = 0; t0 < MAX_T; t0 += KT) {
    // A chunk (f16, global, independent of LDS) - overlaps with staging
    const v8h a0l = *(const v8h*)(pa + t0 + hi * 16);
    const v8h a0h = *(const v8h*)(pa + t0 + hi * 16 + 8);
    const v8h a1l = *(const v8h*)(pa + t0 + 32 + hi * 16);
    const v8h a1h = *(const v8h*)(pa + t0 + 32 + hi * 16 + 8);
    __syncthreads();
    // batched staging: 16 b128 loads, then convert + ds_store pass
    float4 raw[16];
#pragma unroll
    for (int it = 0; it < 16; ++it) {
      const int i = it * 256 + threadIdx.x;
      const int g = i & 63, t = i >> 6;
      const int tt = t0 + t;
      const float* src = (tt < START)
          ? kvcache + ((size_t)b * MAX_T + tt) * KV_RANK + n0b + g * 4
          : kvnew + (size_t)(b * SEQ + (tt - START)) * CF + n0b + g * 4;
      raw[it] = *(const float4*)src;
    }
#pragma unroll
    for (int it = 0; it < 16; ++it) {
      const int i = it * 256 + threadIdx.x;
      const int g = i & 63, t = i >> 6;
      ct[(g * 4 + 0) * CTP + t] = (_Float16)raw[it].x;
      ct[(g * 4 + 1) * CTP + t] = (_Float16)raw[it].y;
      ct[(g * 4 + 2) * CTP + t] = (_Float16)raw[it].z;
      ct[(g * 4 + 3) * CTP + t] = (_Float16)raw[it].w;
    }
    __syncthreads();
    // compute: 2 k-steps x 4 n-frags = 8 wmma
    const v16h a0 = joinA_f16(a0l, a0h);
    const v16h a1 = joinA_f16(a1l, a1h);
#pragma unroll
    for (int nf = 0; nf < 4; ++nf) {
      const _Float16* pb = ct + (size_t)(ng * 64 + nf * 16 + li) * CTP;
      acc[nf] = WMMA(a0, packB_f16(pb, 0, hi), acc[nf]);
      acc[nf] = WMMA(a1, packB_f16(pb, 32, hi), acc[nf]);
    }
  }
#pragma unroll
  for (int nf = 0; nf < 4; ++nf)
#pragma unroll
    for (int r = 0; r < 8; ++r)
      O[((size_t)((b * SEQ + mtile * 16 + hi * 8 + r) * NH + h)) * KV_RANK +
        n0b + ng * 64 + nf * 16 + li] = acc[nf][r];
}

// ---------------------------------------------------------------------------
// K9: per-head value up-proj: O(256x512) x w_up_v[h]^T(512x128) -> Ohead
// ---------------------------------------------------------------------------
__global__ void __launch_bounds__(256)
k_o_up(const float* __restrict__ O, const float* __restrict__ wkvu,
       float* __restrict__ Ohead) {
  const int gw = global_wave();            // 512 waves: 16 h x 16 m x 2 ng
  const int h = gw / 32;
  const int rem = gw % 32;
  const int mtile = rem / 2, ng = rem % 2;
  const int m0 = mtile * 16, n0 = ng * 64;
  gemm16x64(KV_RANK,
    [&](int m) { return O + ((size_t)((m0 + m) * NH + h)) * KV_RANK; },
    [&](int n) {
      return wkvu + (size_t)(h * (D_NR + D_V) + D_NR + n0 + n) * KV_RANK;
    },
    [&](int m, int n, float v) {
      Ohead[(m0 + m) * (NH * D_V) + h * D_V + n0 + n] = v;
    });
}

// ---------------------------------------------------------------------------
// K10: output projection  Ohead(256x2048) x w_out^T(2048x2048) -> out
// ---------------------------------------------------------------------------
__global__ void __launch_bounds__(256)
k_out(const float* __restrict__ Ohead, const float* __restrict__ wout,
      float* __restrict__ out) {
  const int gw = global_wave();            // 512 waves: 16 m x 32 ng
  const int ng = gw % 32, mtile = gw / 32;
  const int m0 = mtile * 16, n0 = ng * 64;
  gemm16x64(DIM,
    [&](int m) { return Ohead + (size_t)(m0 + m) * DIM; },
    [&](int n) { return wout + (size_t)(n0 + n) * DIM; },
    [&](int m, int n, float v) { out[(m0 + m) * DIM + n0 + n] = v; });
}

// ---------------------------------------------------------------------------
extern "C" void kernel_launch(void* const* d_in, const int* in_sizes, int n_in,
                              void* d_out, int out_size, void* d_ws,
                              size_t ws_size, hipStream_t stream) {
  (void)in_sizes; (void)n_in; (void)out_size; (void)ws_size;
  const float* x       = (const float*)d_in[0];
  const float* fc      = (const float*)d_in[2];
  const float* fs      = (const float*)d_in[3];
  const float* kvcache = (const float*)d_in[5];
  const float* krcache = (const float*)d_in[6];
  const float* w_kvd   = (const float*)d_in[7];
  const float* w_qd    = (const float*)d_in[8];
  const float* rmsw    = (const float*)d_in[9];
  const float* w_qu    = (const float*)d_in[10];
  const float* w_kvu   = (const float*)d_in[11];
  const float* w_out   = (const float*)d_in[12];
  float* out = (float*)d_out;

  float* ws = (float*)d_ws;
  float* qdown = ws + WS_QDOWN;
  float* qnorm = ws + WS_QNORM;
  float* q     = ws + WS_Q;
  float* kv    = ws + WS_KV;
  float* qfull = ws + WS_QFULL;
  float* O     = ws + WS_O;
  float* Ohead = ws + WS_OHEAD;
  _Float16* S  = (_Float16*)((char*)d_ws + WS_S16_BYTE);

  k_down_proj<<<42, 256, 0, stream>>>(x, w_qd, w_kvd, qdown, kv);
  k_rms_krope<<<TOK, 256, 0, stream>>>(qdown, rmsw, fc, fs, qnorm, kv);
  k_q_up<<<96, 256, 0, stream>>>(qnorm, w_qu, q);
  k_qrope<<<512, 256, 0, stream>>>(q, fc, fs, qfull);
  k_q_abs<<<256, 256, 0, stream>>>(q, w_kvu, qfull);
  k_scores<<<4096, 128, 0, stream>>>(qfull, kvcache, krcache, kv, S);
  k_softmax<<<4096, 256, 0, stream>>>(S);
  k_pv<<<256, 256, 0, stream>>>(S, kvcache, kv, O);
  k_o_up<<<64, 256, 0, stream>>>(O, w_kvu, Ohead);
  k_out<<<64, 256, 0, stream>>>(Ohead, w_out, out);
}
